// GraphRefiner_52733608460360
// MI455X (gfx1250) — matrix operand order repaired and checked by reading, not compiled, not run on hardware
//
#include <hip/hip_runtime.h>
#include <hip/hip_bf16.h>

#define NROW 8192
#define DIM  512
#define KNN  16

typedef __attribute__((ext_vector_type(16))) __bf16    v16bf;
typedef __attribute__((ext_vector_type(8)))  float     v8f;
typedef __attribute__((ext_vector_type(4)))  float     v4f;
typedef __attribute__((ext_vector_type(4)))  unsigned  v4u;

union FragBF {
    v16bf v;
    v4u   q[2];
    unsigned short u[16];
};

__device__ __forceinline__ unsigned short f2bf(float f) {
    union { float f; unsigned u; } x; x.f = f;
    unsigned r = x.u + 0x7FFFu + ((x.u >> 16) & 1u);   // RNE
    return (unsigned short)(r >> 16);
}

// A fragment (16x32 bf16, row-striped): per lane two contiguous 8-elem chunks.
__device__ __forceinline__ void load_a_frag(FragBF& f, const unsigned short* At,
                                            int n, int half, int kbase) {
    const unsigned short* ap = &At[(size_t)n * DIM + kbase + half * 8];
    f.q[0] = *(const v4u*)ap;
    f.q[1] = *(const v4u*)(ap + 16);
}

// B fragment (32x16 bf16, per lane two 8-elem chunks, same k order as A).
__device__ __forceinline__ void load_b_frag(FragBF& f, const unsigned short* B,
                                            int col, int half, int kbase) {
    const unsigned short* bp = &B[(size_t)col * DIM + kbase + half * 8];
    f.q[0] = *(const v4u*)bp;
    f.q[1] = *(const v4u*)(bp + 16);
}

// Stage a 16x16 d2 tile through LDS (transpose lanes->rows) and merge into
// the per-lane sorted top-16 held by lanes 0..15.
__device__ __forceinline__ void topk_insert_tile(float* d2w, const v8f c, float sqj,
                                                 int jbase, int rowg, int lane,
                                                 int half, int n,
                                                 float (&tv)[16], int (&ti)[16]) {
    #pragma unroll
    for (int v = 0; v < 8; ++v)
        d2w[(v + 8 * half) * 16 + n] = sqj - 2.0f * c[v];
    __builtin_amdgcn_wave_barrier();
    if (lane < 16) {
        #pragma unroll
        for (int cidx = 0; cidx < 16; ++cidx) {
            const float val = d2w[lane * 16 + cidx];
            const int j = jbase + cidx;
            if (j != rowg && val < tv[15]) {
                tv[15] = val; ti[15] = j;
                #pragma unroll
                for (int s2 = 15; s2 > 0; --s2) {
                    if (tv[s2] < tv[s2 - 1]) {
                        float tf = tv[s2]; tv[s2] = tv[s2-1]; tv[s2-1] = tf;
                        int   tt = ti[s2]; ti[s2] = ti[s2-1]; ti[s2-1] = tt;
                    }
                }
            }
        }
    }
    __builtin_amdgcn_wave_barrier();
}

// ---------------------------------------------------------------------------
// Kernel 1: Fused = LN(Z + beta*Y); write f32 to d_out, bf16 to ws, sq-norms.
// ---------------------------------------------------------------------------
__global__ __launch_bounds__(256) void k_ln1(const float* __restrict__ Z,
                                             const float* __restrict__ Y,
                                             const float* __restrict__ g1,
                                             const float* __restrict__ b1,
                                             const float* __restrict__ beta_p,
                                             float* __restrict__ Fused,
                                             unsigned short* __restrict__ Fb,
                                             float* __restrict__ sq) {
    const int r = blockIdx.x, t = threadIdx.x;
    const float beta = beta_p[0];
    const size_t base = (size_t)r * DIM;
    float x0 = Z[base + t]       + beta * Y[base + t];
    float x1 = Z[base + 256 + t] + beta * Y[base + 256 + t];

    __shared__ float red[8];
    float s = x0 + x1;
    #pragma unroll
    for (int o = 16; o > 0; o >>= 1) s += __shfl_xor(s, o);
    if ((t & 31) == 0) red[t >> 5] = s;
    __syncthreads();
    float tot = red[0]+red[1]+red[2]+red[3]+red[4]+red[5]+red[6]+red[7];
    const float mean = tot * (1.0f / DIM);
    __syncthreads();
    float ss = x0*x0 + x1*x1;
    #pragma unroll
    for (int o = 16; o > 0; o >>= 1) ss += __shfl_xor(ss, o);
    if ((t & 31) == 0) red[t >> 5] = ss;
    __syncthreads();
    float tot2 = red[0]+red[1]+red[2]+red[3]+red[4]+red[5]+red[6]+red[7];
    const float var = tot2 * (1.0f / DIM) - mean * mean;
    const float inv = rsqrtf(var + 1e-5f);
    __syncthreads();

    float f0 = (x0 - mean) * inv * g1[t]       + b1[t];
    float f1 = (x1 - mean) * inv * g1[t + 256] + b1[t + 256];
    Fused[base + t]       = f0;
    Fused[base + 256 + t] = f1;
    Fb[base + t]          = f2bf(f0);
    Fb[base + 256 + t]    = f2bf(f1);

    float s2 = f0*f0 + f1*f1;
    #pragma unroll
    for (int o = 16; o > 0; o >>= 1) s2 += __shfl_xor(s2, o);
    if ((t & 31) == 0) red[t >> 5] = s2;
    __syncthreads();
    if (t == 0)
        sq[r] = red[0]+red[1]+red[2]+red[3]+red[4]+red[5]+red[6]+red[7];
}

// ---------------------------------------------------------------------------
// Kernel 2: W f32 -> bf16
// ---------------------------------------------------------------------------
__global__ __launch_bounds__(256) void k_wconv(const float* __restrict__ W,
                                               unsigned short* __restrict__ Wb) {
    int i = blockIdx.x * 256 + threadIdx.x;
    Wb[i] = f2bf(W[i]);
}

// ---------------------------------------------------------------------------
// Kernel 3: WMMA Gram strip + streaming per-row top-16 + graph strip write.
// Block = 512 threads (16 waves) owns 16 rows. Each wave processes 32 column
// tiles (two per iteration, shared A fragment, double-buffered loads).
// ---------------------------------------------------------------------------
__global__ __launch_bounds__(512) void k_knn(const unsigned short* __restrict__ Fb,
                                             const float* __restrict__ sq,
                                             float* __restrict__ graph,
                                             int* __restrict__ knn_idx) {
    __shared__ __align__(16) char smem[49152];
    unsigned short* Atile = (unsigned short*)smem;          // 16KB, phase 1
    float*          valb  = (float*)smem;                   // 16KB, phase 2 (reuses Atile)
    float*          d2s   = (float*)(smem + 16384);         // 16KB, phase 1 scratch
    int*            idxb  = (int*)(smem + 32768);           // 16KB, phase 2

    const int tid  = threadIdx.x;
    const int wave = tid >> 5;
    const int lane = tid & 31;
    const int half = lane >> 4;
    const int n    = lane & 15;
    const int rb   = blockIdx.x * 16;
    const float FLT_INF = __builtin_huge_valf();
    float* d2w = d2s + (size_t)wave * 256;

    // cooperative load of 16-row A strip (16x512 bf16)
    {
        const int e = tid * 16, row = e >> 9, col = e & 511;
        const v4u* s = (const v4u*)&Fb[(size_t)(rb + row) * DIM + col];
        v4u* d = (v4u*)&Atile[e];
        d[0] = s[0]; d[1] = s[1];
    }
    __syncthreads();

    float tv[16]; int ti[16];
    #pragma unroll
    for (int i = 0; i < 16; ++i) { tv[i] = FLT_INF; ti[i] = -1; }
    const int rowg = rb + lane;   // meaningful for lanes 0..15

    for (int t = 0; t < 16; ++t) {
        const int j0 = (wave + (2 * t)     * 16) * 16;
        const int j1 = (wave + (2 * t + 1) * 16) * 16;
        const float sqj0 = sq[j0 + n];
        const float sqj1 = sq[j1 + n];

        v8f c0 = {0.f,0.f,0.f,0.f,0.f,0.f,0.f,0.f};
        v8f c1 = {0.f,0.f,0.f,0.f,0.f,0.f,0.f,0.f};
        FragBF fa[2], fb0[2], fb1[2];
        load_a_frag(fa[0],  Atile, n, half, 0);
        load_b_frag(fb0[0], Fb, j0 + n, half, 0);
        load_b_frag(fb1[0], Fb, j1 + n, half, 0);
        #pragma unroll
        for (int ks = 0; ks < 16; ++ks) {
            const int cur = ks & 1, nxt = cur ^ 1;
            if (ks < 15) {   // prefetch next k-step while current WMMAs run
                load_a_frag(fa[nxt],  Atile, n, half, (ks + 1) * 32);
                load_b_frag(fb0[nxt], Fb, j0 + n, half, (ks + 1) * 32);
                load_b_frag(fb1[nxt], Fb, j1 + n, half, (ks + 1) * 32);
            }
            c0 = __builtin_amdgcn_wmma_f32_16x16x32_bf16(
                     false, fa[cur].v, false, fb0[cur].v, (short)0, c0, false, false);
            c1 = __builtin_amdgcn_wmma_f32_16x16x32_bf16(
                     false, fa[cur].v, false, fb1[cur].v, (short)0, c1, false, false);
        }
        topk_insert_tile(d2w, c0, sqj0, j0, rowg, lane, half, n, tv, ti);
        topk_insert_tile(d2w, c1, sqj1, j1, rowg, lane, half, n, tv, ti);
    }

    __syncthreads();   // Atile dead -> region becomes valb
    if (lane < 16) {
        #pragma unroll
        for (int i = 0; i < 16; ++i) {
            valb[((size_t)wave * 16 + lane) * 16 + i] = tv[i];
            idxb[((size_t)wave * 16 + lane) * 16 + i] = ti[i];
        }
    }
    __syncthreads();

    // cross-wave merge: wave w selects final top-16 for strip row w.
    const int row = wave;
    float mv[8]; int mi[8];
    #pragma unroll
    for (int u = 0; u < 8; ++u) {
        const int cnd = lane * 8 + u;
        mv[u] = valb[((size_t)(cnd >> 4) * 16 + row) * 16 + (cnd & 15)];
        mi[u] = idxb[((size_t)(cnd >> 4) * 16 + row) * 16 + (cnd & 15)];
    }
    #pragma unroll
    for (int i = 0; i < 8; ++i)
        #pragma unroll
        for (int jx = 0; jx < 7; ++jx)
            if (mv[jx] > mv[jx + 1]) {
                float tf = mv[jx]; mv[jx] = mv[jx+1]; mv[jx+1] = tf;
                int   tt = mi[jx]; mi[jx] = mi[jx+1]; mi[jx+1] = tt;
            }

    float cur = mv[0]; int curi = mi[0];
    int myidx = -1;
    #pragma unroll
    for (int r16 = 0; r16 < 16; ++r16) {
        float bv = cur; int bl = lane;
        #pragma unroll
        for (int o = 16; o > 0; o >>= 1) {
            float ov = __shfl_xor(bv, o);
            int   ol = __shfl_xor(bl, o);
            if (ov < bv || (ov == bv && ol < bl)) { bv = ov; bl = ol; }
        }
        const int widx = __shfl(curi, bl);
        if (lane == r16) myidx = widx;
        if (lane == bl) {
            mv[0]=mv[1]; mi[0]=mi[1]; mv[1]=mv[2]; mi[1]=mi[2];
            mv[2]=mv[3]; mi[2]=mi[3]; mv[3]=mv[4]; mi[3]=mi[4];
            mv[4]=mv[5]; mi[4]=mi[5]; mv[5]=mv[6]; mi[5]=mi[6];
            mv[6]=mv[7]; mi[6]=mi[7]; mv[7] = FLT_INF; mi[7] = -1;
            cur = mv[0]; curi = mi[0];
        }
    }

    const int outrow = rb + row;
    if (lane < 16) knn_idx[(size_t)outrow * 16 + lane] = myidx;

    // dense adjacency row: vectorized zero-fill, then scatter 1/16 entries.
    // Same-wave vector stores complete in order; wait makes it explicit.
    float* grow = graph + (size_t)outrow * NROW;
    const v4f z4 = {0.f, 0.f, 0.f, 0.f};
    #pragma unroll 4
    for (int c = lane * 4; c < NROW; c += 128)
        *(v4f*)&grow[c] = z4;
    asm volatile("s_wait_storecnt 0x0" ::: "memory");
    if (lane < 16 && myidx >= 0)
        grow[myidx] = 1.0f / 16.0f;
}

// ---------------------------------------------------------------------------
// Kernel 4: propagated = mean of 16 neighbor rows of Fused; emit bf16.
// ---------------------------------------------------------------------------
__global__ __launch_bounds__(256) void k_prop(const float* __restrict__ Fused,
                                              const int* __restrict__ knn_idx,
                                              unsigned short* __restrict__ Pb) {
    const int r = blockIdx.x, t = threadIdx.x;
    __shared__ int nb[16];
    if (t < 16) nb[t] = knn_idx[(size_t)r * 16 + t];
    __syncthreads();
    float a0 = 0.f, a1 = 0.f;
    #pragma unroll
    for (int i = 0; i < 16; ++i) {
        const float* fr = Fused + (size_t)nb[i] * DIM;
        a0 += fr[t];
        a1 += fr[t + 256];
    }
    const size_t base = (size_t)r * DIM;
    Pb[base + t]       = f2bf(a0 * (1.0f / 16.0f));
    Pb[base + 256 + t] = f2bf(a1 * (1.0f / 16.0f));
}

// ---------------------------------------------------------------------------
// Kernel 5: hidden = LN(P @ W^T + bias). WMMA GEMM + fused LN in LDS.
// Block = 512 threads (16 waves) computes 16 full output rows.
// ---------------------------------------------------------------------------
__global__ __launch_bounds__(512) void k_hidden(const unsigned short* __restrict__ Pb,
                                                const unsigned short* __restrict__ Wb,
                                                const float* __restrict__ bias,
                                                const float* __restrict__ g2,
                                                const float* __restrict__ b2,
                                                float* __restrict__ hidden) {
    __shared__ __align__(16) unsigned short Ptile[16 * DIM];  // 16KB
    __shared__ float Htile[16 * DIM];                         // 32KB

    const int tid  = threadIdx.x;
    const int wave = tid >> 5;
    const int lane = tid & 31;
    const int half = lane >> 4;
    const int n    = lane & 15;
    const int rb   = blockIdx.x * 16;

    {
        const int e = tid * 16, row = e >> 9, col = e & 511;
        const v4u* s = (const v4u*)&Pb[(size_t)(rb + row) * DIM + col];
        v4u* d = (v4u*)&Ptile[e];
        d[0] = s[0]; d[1] = s[1];
    }
    __syncthreads();

    const int o = wave * 16 + n;
    v8f c = {0.f,0.f,0.f,0.f,0.f,0.f,0.f,0.f};
    FragBF fa[2], fb[2];
    load_a_frag(fa[0], Ptile, n, half, 0);
    load_b_frag(fb[0], Wb, o, half, 0);
    #pragma unroll
    for (int ks = 0; ks < 16; ++ks) {
        const int cur = ks & 1, nxt = cur ^ 1;
        if (ks < 15) {
            load_a_frag(fa[nxt], Ptile, n, half, (ks + 1) * 32);
            load_b_frag(fb[nxt], Wb, o, half, (ks + 1) * 32);
        }
        c = __builtin_amdgcn_wmma_f32_16x16x32_bf16(
                false, fa[cur].v, false, fb[cur].v, (short)0, c, false, false);
    }
    const float bs = bias[o];
    #pragma unroll
    for (int v = 0; v < 8; ++v)
        Htile[(size_t)(v + 8 * half) * DIM + o] = c[v] + bs;
    __syncthreads();

    const int row = wave;
    float vals[16];
    float s = 0.f, ss = 0.f;
    #pragma unroll
    for (int t16 = 0; t16 < 16; ++t16) {
        const float v = Htile[(size_t)row * DIM + lane + 32 * t16];
        vals[t16] = v; s += v; ss += v * v;
    }
    #pragma unroll
    for (int of = 16; of > 0; of >>= 1) { s += __shfl_xor(s, of); ss += __shfl_xor(ss, of); }
    const float mean = s * (1.0f / DIM);
    const float var  = ss * (1.0f / DIM) - mean * mean;
    const float inv  = rsqrtf(var + 1e-5f);
    #pragma unroll
    for (int t16 = 0; t16 < 16; ++t16) {
        const int col = lane + 32 * t16;
        hidden[(size_t)(rb + row) * DIM + col] = (vals[t16] - mean) * inv * g2[col] + b2[col];
    }
}

// ---------------------------------------------------------------------------
extern "C" void kernel_launch(void* const* d_in, const int* in_sizes, int n_in,
                              void* d_out, int out_size, void* d_ws, size_t ws_size,
                              hipStream_t stream) {
    const float* Z      = (const float*)d_in[0];
    const float* Y      = (const float*)d_in[1];
    const float* ln1_g  = (const float*)d_in[2];
    const float* ln1_b  = (const float*)d_in[3];
    const float* W      = (const float*)d_in[4];
    const float* bias   = (const float*)d_in[5];
    const float* ln2_g  = (const float*)d_in[6];
    const float* ln2_b  = (const float*)d_in[7];
    /* d_in[8] = graph_k (always 16 here), d_in[9] = beta */
    const float* beta_p = (const float*)d_in[9];

    float* out_fused  = (float*)d_out;
    float* out_graph  = out_fused + (size_t)NROW * DIM;
    float* out_hidden = out_graph + (size_t)NROW * NROW;

    char* ws = (char*)d_ws;
    unsigned short* Fb  = (unsigned short*)(ws);                         //  8 MB
    unsigned short* Pb  = (unsigned short*)(ws + (size_t)NROW*DIM*2);    //  8 MB
    unsigned short* Wb  = (unsigned short*)(ws + (size_t)NROW*DIM*4);    // 512 KB
    float*          sqn = (float*)(ws + (size_t)NROW*DIM*4 + DIM*DIM*2); // 32 KB
    int*            kix = (int*)((char*)sqn + NROW*4);                   // 512 KB

    k_ln1 <<<NROW, 256, 0, stream>>>(Z, Y, ln1_g, ln1_b, beta_p, out_fused, Fb, sqn);
    k_wconv<<<(DIM * DIM) / 256, 256, 0, stream>>>(W, Wb);
    k_knn <<<NROW / 16, 512, 0, stream>>>(Fb, sqn, out_graph, kix);
    k_prop<<<NROW, 256, 0, stream>>>(out_fused, kix, Pb);
    k_hidden<<<NROW / 16, 512, 0, stream>>>(Pb, Wb, bias, ln2_g, ln2_b, out_hidden);
}
